// BiDirectionalMinGRU_51333449122327
// MI455X (gfx1250) — matrix operand order repaired and checked by reading, not compiled, not run on hardware
//
#include <hip/hip_runtime.h>
#include <hip/hip_bf16.h>
#include <math.h>

// ---------------------------------------------------------------------------
// BiDirectional minGRU for MI455X (gfx1250): bf16 WMMA GEMMs + channel scan
// B=8, L=4096, H=512, T=8 (fixed by the reference)
// ---------------------------------------------------------------------------

typedef __attribute__((ext_vector_type(8)))  __bf16 v8bf;
typedef __attribute__((ext_vector_type(16))) __bf16 v16bf;
typedef __attribute__((ext_vector_type(8)))  float  v8f;

#define BB 8
#define LL 4096
#define HH 512
#define MTOT (BB * LL)          // 32768 rows
#define KDIM 512

// ---------------------------------------------------------------------------
// Kernel 0: weight transpose + f32->bf16 convert.  dst[n*512+k] = src[k*512+n]
// ---------------------------------------------------------------------------
__global__ __launch_bounds__(256)
void wt_transpose_bf16(const float* __restrict__ src, __bf16* __restrict__ dst) {
    int k = blockIdx.x * 16 + (threadIdx.x & 15);
    int n = blockIdx.y * 16 + (threadIdx.x >> 4);
    dst[(size_t)n * KDIM + k] = (__bf16)src[(size_t)k * HH + n];
}

// ---------------------------------------------------------------------------
// Kernel 1: time-MLP + input projections (K=10), writes t_enc to d_out and
// bf16 xp_f / xp_b (backward time-flipped) to workspace.
// One block per (b,l); 512 threads, one per hidden channel.
// ---------------------------------------------------------------------------
__global__ __launch_bounds__(512)
void proj_kernel(const float* __restrict__ x,  const float* __restrict__ t,
                 const float* __restrict__ te_w1, const float* __restrict__ te_b1,
                 const float* __restrict__ te_w2, const float* __restrict__ te_b2,
                 const float* __restrict__ fw, const float* __restrict__ fb,
                 const float* __restrict__ bw, const float* __restrict__ bb,
                 float* __restrict__ tenc_out,
                 __bf16* __restrict__ xp_f, __bf16* __restrict__ xp_b) {
    const int bl = blockIdx.x;            // 0..B*L-1
    const int b  = bl >> 12;              // /4096
    const int l  = bl & (LL - 1);
    const int h  = threadIdx.x;

    const float ts = t[bl] - t[(size_t)b * LL];

    float r[8];
#pragma unroll
    for (int j = 0; j < 8; ++j) {
        float v = ts * te_w1[j] + te_b1[j];
        r[j] = v > 0.f ? v : 0.f;
    }
    float xc[10];
    xc[0] = x[(size_t)bl * 2 + 0];
    xc[1] = x[(size_t)bl * 2 + 1];
#pragma unroll
    for (int i = 0; i < 8; ++i) {
        float e = te_b2[i];
#pragma unroll
        for (int j = 0; j < 8; ++j) e = fmaf(r[j], te_w2[j * 8 + i], e);
        xc[2 + i] = e;
    }
    if (h < 8) tenc_out[(size_t)bl * 8 + h] = xc[2 + h];

    float vf = fb[h];
    float vb = bb[h];
#pragma unroll
    for (int k = 0; k < 10; ++k) {
        vf = fmaf(xc[k], fw[(size_t)k * HH + h], vf);
        vb = fmaf(xc[k], bw[(size_t)k * HH + h], vb);
    }
    xp_f[(size_t)bl * HH + h] = (__bf16)vf;
    const int lflip = LL - 1 - l;
    xp_b[((size_t)b * LL + lflip) * HH + h] = (__bf16)vb;
}

// ---------------------------------------------------------------------------
// WMMA fragment loaders.
// A (16x32 bf16, row-major, ld = 512):
//   lane (0..31): row = lane&15; kgrp = (lane>>4)*8
//   elems 0..7  -> K = k0+kgrp+0..7      (contig 16B)
//   elems 8..15 -> K = k0+16+kgrp+0..7   (contig 16B)
// B (32x16 bf16 from N-major W^T, ld = 512):
//   lane: col n = lane&15; kgrp = (lane>>4)*8; elems 0..7 -> K = k0+kgrp+0..7, elems 8..15 -> K = k0+16+kgrp+0..7
// ---------------------------------------------------------------------------
__device__ __forceinline__ v16bf load_frag_a(const __bf16* p) {
    v8bf lo = *(const v8bf*)p;
    v8bf hi = *(const v8bf*)(p + 16);
    return __builtin_shufflevector(lo, hi, 0,1,2,3,4,5,6,7,8,9,10,11,12,13,14,15);
}
__device__ __forceinline__ v16bf load_frag_b(const __bf16* p) {
    v8bf lo = *(const v8bf*)p;
    v8bf hi = *(const v8bf*)(p + 16);
    return __builtin_shufflevector(lo, hi, 0,1,2,3,4,5,6,7,8,9,10,11,12,13,14,15);
}

// ---------------------------------------------------------------------------
// Kernel 2: dual GEMM + fused gate epilogue.
//   z_pre = xp @ Wz + bz ; h_pre = xp @ Wh + bh
//   a = 1 - sigmoid(z_pre) ; g = sigmoid(z_pre) * h_pre
// Block = 8 waves; wave tile = 32(M) x 32(N); block tile = 128 x 64.
// grid = (N/64 = 8, M/128 = 256)
// ---------------------------------------------------------------------------
__global__ __launch_bounds__(256)
void mingru_gemm(const __bf16* __restrict__ xp,
                 const __bf16* __restrict__ wzT, const __bf16* __restrict__ whT,
                 const float* __restrict__ bz,  const float* __restrict__ bh,
                 float* __restrict__ a_out, float* __restrict__ g_out) {
    const int lane = threadIdx.x & 31;
    const int wave = threadIdx.x >> 5;
    const int mw   = (wave >> 1) * 32;          // 0,32,64,96
    const int nw   = (wave & 1) * 32;           // 0,32
    const int mblk = blockIdx.y * 128;
    const int nblk = blockIdx.x * 64;

    const int lrow = lane & 15;
    const int kgA  = (lane >> 4) << 3;          // A: 0 or 8
    const int kgB  = (lane >> 4) << 3;          // B: 0 or 8

    v8f accz[2][2] = {};
    v8f acch[2][2] = {};

    const __bf16* aBase = xp  + (size_t)(mblk + mw + lrow) * KDIM + kgA;
    const __bf16* zBase = wzT + (size_t)(nblk + nw + lrow) * KDIM + kgB;
    const __bf16* hBase = whT + (size_t)(nblk + nw + lrow) * KDIM + kgB;

#pragma unroll 2
    for (int k0 = 0; k0 < KDIM; k0 += 32) {
        v16bf a0 = load_frag_a(aBase + k0);
        v16bf a1 = load_frag_a(aBase + (size_t)16 * KDIM + k0);
        v16bf z0 = load_frag_b(zBase + k0);
        v16bf z1 = load_frag_b(zBase + (size_t)16 * KDIM + k0);
        v16bf h0 = load_frag_b(hBase + k0);
        v16bf h1 = load_frag_b(hBase + (size_t)16 * KDIM + k0);

        accz[0][0] = __builtin_amdgcn_wmma_f32_16x16x32_bf16(false, a0, false, z0, (short)0, accz[0][0], false, false);
        accz[0][1] = __builtin_amdgcn_wmma_f32_16x16x32_bf16(false, a0, false, z1, (short)0, accz[0][1], false, false);
        accz[1][0] = __builtin_amdgcn_wmma_f32_16x16x32_bf16(false, a1, false, z0, (short)0, accz[1][0], false, false);
        accz[1][1] = __builtin_amdgcn_wmma_f32_16x16x32_bf16(false, a1, false, z1, (short)0, accz[1][1], false, false);
        acch[0][0] = __builtin_amdgcn_wmma_f32_16x16x32_bf16(false, a0, false, h0, (short)0, acch[0][0], false, false);
        acch[0][1] = __builtin_amdgcn_wmma_f32_16x16x32_bf16(false, a0, false, h1, (short)0, acch[0][1], false, false);
        acch[1][0] = __builtin_amdgcn_wmma_f32_16x16x32_bf16(false, a1, false, h0, (short)0, acch[1][0], false, false);
        acch[1][1] = __builtin_amdgcn_wmma_f32_16x16x32_bf16(false, a1, false, h1, (short)0, acch[1][1], false, false);
    }

    // Epilogue: C/D layout -> lane l: n = base + (l&15); VGPR r: m = base + 8*(l>>4) + r
    const int lhi  = lane >> 4;
    const int lcol = lane & 15;
#pragma unroll
    for (int i = 0; i < 2; ++i) {
#pragma unroll
        for (int j = 0; j < 2; ++j) {
            const int n   = nblk + nw + 16 * j + lcol;
            const float bzv = bz[n];
            const float bhv = bh[n];
#pragma unroll
            for (int r = 0; r < 8; ++r) {
                const int m = mblk + mw + 16 * i + 8 * lhi + r;
                const float zp = accz[i][j][r] + bzv;
                const float z  = 1.f / (1.f + __expf(-zp));
                const float ht = acch[i][j][r] + bhv;
                const size_t idx = (size_t)m * HH + n;
                a_out[idx] = 1.f - z;
                g_out[idx] = z * ht;
            }
        }
    }
}

// ---------------------------------------------------------------------------
// Kernel 3: sequential recurrence per (dir, b, h) channel.
//   H_t = a_t * H_{t-1} + g_t  with one-step prefetch.
// Forward: h_fwd[b,0]=0, h_fwd[b,t+1]=H_t.
// Backward (arrays stored time-flipped, s = L-1-t):
//   h_bwd[b,L-1]=0, S[s] -> h_bwd[b, L-2-s].
// 8192 threads total (256 wave32 waves).
// ---------------------------------------------------------------------------
__global__ __launch_bounds__(256)
void scan_kernel(const float* __restrict__ a_f, const float* __restrict__ g_f,
                 const float* __restrict__ a_b, const float* __restrict__ g_b,
                 float* __restrict__ out) {
    const int tid = blockIdx.x * blockDim.x + threadIdx.x;
    const int dir = tid >> 12;          // 4096 channels per direction
    const int c   = tid & 4095;
    const int b   = c >> 9;
    const int h   = c & 511;

    const float* A = dir ? a_b : a_f;
    const float* G = dir ? g_b : g_f;
    float* o = out + (size_t)BB * LL * 8 + (size_t)dir * BB * LL * HH;

    const size_t base = (size_t)b * LL * HH + h;
    float hc = 0.f;

    if (dir == 0) {
        o[base] = 0.f;                              // t = 0
        float an = A[base], gn = G[base];
        for (int tt = 0; tt < LL - 1; ++tt) {
            const float ac = an, gc = gn;
            const size_t nxt = base + (size_t)(tt + 1) * HH;
            an = A[nxt]; gn = G[nxt];               // prefetch next step
            hc = fmaf(ac, hc, gc);
            o[nxt] = hc;                            // h_fwd[b, tt+1]
        }
    } else {
        o[base + (size_t)(LL - 1) * HH] = 0.f;      // t = L-1
        float an = A[base], gn = G[base];
        for (int s = 0; s < LL - 1; ++s) {
            const float ac = an, gc = gn;
            const size_t nxt = base + (size_t)(s + 1) * HH;
            an = A[nxt]; gn = G[nxt];
            hc = fmaf(ac, hc, gc);
            o[base + (size_t)(LL - 2 - s) * HH] = hc;  // h_bwd[b, L-2-s]
        }
    }
}

// ---------------------------------------------------------------------------
// Launcher
// ---------------------------------------------------------------------------
extern "C" void kernel_launch(void* const* d_in, const int* in_sizes, int n_in,
                              void* d_out, int out_size, void* d_ws, size_t ws_size,
                              hipStream_t stream) {
    const float* x      = (const float*)d_in[0];
    const float* t      = (const float*)d_in[1];
    const float* te_w1  = (const float*)d_in[2];
    const float* te_b1  = (const float*)d_in[3];
    const float* te_w2  = (const float*)d_in[4];
    const float* te_b2  = (const float*)d_in[5];
    const float* fproj_w= (const float*)d_in[6];
    const float* fproj_b= (const float*)d_in[7];
    const float* bproj_w= (const float*)d_in[8];
    const float* bproj_b= (const float*)d_in[9];
    const float* f_wz   = (const float*)d_in[10];
    const float* f_bz   = (const float*)d_in[11];
    const float* f_wh   = (const float*)d_in[12];
    const float* f_bh   = (const float*)d_in[13];
    const float* b_wz   = (const float*)d_in[14];
    const float* b_bz   = (const float*)d_in[15];
    const float* b_wh   = (const float*)d_in[16];
    const float* b_bh   = (const float*)d_in[17];

    float* out = (float*)d_out;

    // Workspace layout
    char* ws = (char*)d_ws;
    const size_t xpBytes = (size_t)MTOT * HH * sizeof(__bf16);   // 32 MiB each
    const size_t wBytes  = (size_t)KDIM * HH * sizeof(__bf16);   // 512 KiB each
    const size_t agBytes = (size_t)MTOT * HH * sizeof(float);    // 64 MiB each

    __bf16* xp_f  = (__bf16*)(ws);
    __bf16* xp_b  = (__bf16*)(ws + xpBytes);
    __bf16* wzT_f = (__bf16*)(ws + 2 * xpBytes);
    __bf16* whT_f = (__bf16*)(ws + 2 * xpBytes + wBytes);
    __bf16* wzT_b = (__bf16*)(ws + 2 * xpBytes + 2 * wBytes);
    __bf16* whT_b = (__bf16*)(ws + 2 * xpBytes + 3 * wBytes);
    float*  a_f   = (float*)(ws + 2 * xpBytes + 4 * wBytes);
    float*  g_f   = (float*)(ws + 2 * xpBytes + 4 * wBytes + agBytes);
    float*  a_b   = (float*)(ws + 2 * xpBytes + 4 * wBytes + 2 * agBytes);
    float*  g_b   = (float*)(ws + 2 * xpBytes + 4 * wBytes + 3 * agBytes);

    // 0) weight transpose+convert (512x512 each)
    {
        dim3 grid(KDIM / 16, HH / 16);
        wt_transpose_bf16<<<grid, 256, 0, stream>>>(f_wz, wzT_f);
        wt_transpose_bf16<<<grid, 256, 0, stream>>>(f_wh, whT_f);
        wt_transpose_bf16<<<grid, 256, 0, stream>>>(b_wz, wzT_b);
        wt_transpose_bf16<<<grid, 256, 0, stream>>>(b_wh, whT_b);
    }

    // 1) time-MLP + projections (t_enc -> d_out, xp -> ws)
    proj_kernel<<<MTOT, 512, 0, stream>>>(x, t, te_w1, te_b1, te_w2, te_b2,
                                          fproj_w, fproj_b, bproj_w, bproj_b,
                                          out, xp_f, xp_b);

    // 2) dual WMMA GEMM per direction -> a, g
    {
        dim3 grid(HH / 64, MTOT / 128);
        mingru_gemm<<<grid, 256, 0, stream>>>(xp_f, wzT_f, whT_f, f_bz, f_bh, a_f, g_f);
        mingru_gemm<<<grid, 256, 0, stream>>>(xp_b, wzT_b, whT_b, b_bz, b_bh, a_b, g_b);
    }

    // 3) channel-parallel recurrence, both directions in one launch
    scan_kernel<<<(2 * BB * HH) / 256, 256, 0, stream>>>(a_f, g_f, a_b, g_b, out);
}
